// GNNDetector_29085518529193
// MI455X (gfx1250) — matrix operand order repaired and checked by reading, not compiled, not run on hardware
//
#include <hip/hip_runtime.h>

typedef __attribute__((ext_vector_type(16))) _Float16 v16h;
typedef __attribute__((ext_vector_type(8)))  float    v8f;

#define N_NODES 50000
#define N_EDGES 800000
#define N_GRAPH 64
#define F_IN    128
#define HDIM    256

// ---------- small utility kernels ----------
__global__ void k_fill(float* p, float v, int n) {
    int i = blockIdx.x * blockDim.x + threadIdx.x;
    if (i < n) p[i] = v;
}

__global__ void k_deg(const int* __restrict__ dst, float* __restrict__ deg, int E) {
    int i = blockIdx.x * blockDim.x + threadIdx.x;
    if (i < E) atomicAdd(&deg[dst[i]], 1.0f);
}

__global__ void k_rsqrt(float* d, int n) {
    int i = blockIdx.x * blockDim.x + threadIdx.x;
    if (i < n) d[i] = rsqrtf(d[i]);
}

__global__ void k_relu(float* p, int n) {
    int i = blockIdx.x * blockDim.x + threadIdx.x;
    if (i < n) p[i] = fmaxf(p[i], 0.0f);
}

// transpose: in[K][Nc] -> out[Nc][K] (weights only, < 2 MB total)
__global__ void k_transpose(const float* __restrict__ in, float* __restrict__ out,
                            int K, int Nc) {
    int i = blockIdx.x * blockDim.x + threadIdx.x;
    if (i >= K * Nc) return;
    int c = i / K;
    int k = i - c * K;
    out[i] = in[(size_t)k * Nc + c];   // coalesced write, strided read
}

// Build a v16h fragment from two runs of 8 consecutive floats (4x b128 loads).
__device__ __forceinline__ v16h frag16(const float* __restrict__ p0,
                                       const float* __restrict__ p1) {
    float4 x0 = ((const float4*)p0)[0];
    float4 x1 = ((const float4*)p0)[1];
    float4 x2 = ((const float4*)p1)[0];
    float4 x3 = ((const float4*)p1)[1];
    v16h f;
    f[0]  = (_Float16)x0.x; f[1]  = (_Float16)x0.y;
    f[2]  = (_Float16)x0.z; f[3]  = (_Float16)x0.w;
    f[4]  = (_Float16)x1.x; f[5]  = (_Float16)x1.y;
    f[6]  = (_Float16)x1.z; f[7]  = (_Float16)x1.w;
    f[8]  = (_Float16)x2.x; f[9]  = (_Float16)x2.y;
    f[10] = (_Float16)x2.z; f[11] = (_Float16)x2.w;
    f[12] = (_Float16)x3.x; f[13] = (_Float16)x3.y;
    f[14] = (_Float16)x3.z; f[15] = (_Float16)x3.w;
    return f;
}

// ---------- WMMA GEMM: C[M,Nc] = A[M,K] @ Bt^T  (Bt is [Nc,K], pre-transposed) --
// One wave32 per 16x64 strip: A fragment loaded once per K-step, reused by 4
// B fragments -> 4x v_wmma_f32_16x16x32_f16 per iteration. All fragment loads
// are contiguous 16-float runs per lane => global_load_b128.
// M mult of 16, K mult of 32, Nc mult of 64.
template<bool BIAS, bool RELU>
__global__ void __launch_bounds__(128)
k_gemm_wmma(const float* __restrict__ A, const float* __restrict__ Bt,
            float* __restrict__ C, int M, int K, int Nc,
            const float* __restrict__ bias)
{
    const int stripsN = Nc >> 6;
    const int strip   = blockIdx.x * blockDim.y + threadIdx.y;
    const int total   = (M >> 4) * stripsN;
    if (strip >= total) return;

    const int tm = strip / stripsN;
    const int tn = strip - tm * stripsN;
    const int m0 = tm << 4, n0 = tn << 6;

    const int lane   = threadIdx.x;   // 0..31
    const int laneHi = lane >> 4;     // 0 or 1
    const int l16    = lane & 15;

    v8f c0 = {}, c1 = {}, c2 = {}, c3 = {};
    const float* __restrict__ Arow = A + (size_t)(m0 + l16) * K;
    // Bt rows for the 4 N-tiles this wave covers (column l16 of each tile)
    const float* __restrict__ B0 = Bt + (size_t)(n0 + l16)      * K;
    const float* __restrict__ B1 = Bt + (size_t)(n0 + 16 + l16) * K;
    const float* __restrict__ B2 = Bt + (size_t)(n0 + 32 + l16) * K;
    const float* __restrict__ B3 = Bt + (size_t)(n0 + 48 + l16) * K;

    for (int kb = 0; kb < K; kb += 32) {
        // A 16x32 fragment: lane-half selects K 16-group
        //   elements 0..15 -> K = kb + 16*laneHi + 0..15
        v16h a = frag16(Arow + kb + (laneHi << 4),
                        Arow + kb + (laneHi << 4) + 8);
        // B 32x16 fragment: lane = column, lane-half selects K 16-group
        //   elements 0..15 -> K = kb + 16*laneHi + 0..15  (contiguous in Bt row)
        int ko = kb + (laneHi << 4);
        v16h b0 = frag16(B0 + ko, B0 + ko + 8);
        v16h b1 = frag16(B1 + ko, B1 + ko + 8);
        v16h b2 = frag16(B2 + ko, B2 + ko + 8);
        v16h b3 = frag16(B3 + ko, B3 + ko + 8);

        c0 = __builtin_amdgcn_wmma_f32_16x16x32_f16(false, a, false, b0, (short)0, c0, false, false);
        c1 = __builtin_amdgcn_wmma_f32_16x16x32_f16(false, a, false, b1, (short)0, c1, false, false);
        c2 = __builtin_amdgcn_wmma_f32_16x16x32_f16(false, a, false, b2, (short)0, c2, false, false);
        c3 = __builtin_amdgcn_wmma_f32_16x16x32_f16(false, a, false, b3, (short)0, c3, false, false);
    }

    // Epilogue: straight-line
    float bv0 = 0.f, bv1 = 0.f, bv2 = 0.f, bv3 = 0.f;
    if (BIAS) {
        const float* __restrict__ bp = bias + n0 + l16;
        bv0 = bp[0]; bv1 = bp[16]; bv2 = bp[32]; bv3 = bp[48];
    }
#pragma unroll
    for (int r = 0; r < 8; ++r) {
        int row = m0 + r + (laneHi << 3);
        float* __restrict__ Crow = C + (size_t)row * Nc + n0 + l16;
        float v0 = c0[r] + bv0;
        float v1 = c1[r] + bv1;
        float v2 = c2[r] + bv2;
        float v3 = c3[r] + bv3;
        if (RELU) {
            v0 = fmaxf(v0, 0.f); v1 = fmaxf(v1, 0.f);
            v2 = fmaxf(v2, 0.f); v3 = fmaxf(v3, 0.f);
        }
        Crow[0]  = v0;
        Crow[16] = v1;
        Crow[32] = v2;
        Crow[48] = v3;
    }
}

// ---------- agg init: out = h * dis^2 + bias (self-loop term) ----------
__global__ void k_init_agg(const float* __restrict__ h, const float* __restrict__ dis,
                           const float* __restrict__ bias, float* __restrict__ outp,
                           int total, int H)
{
    int i = blockIdx.x * blockDim.x + threadIdx.x;
    if (i >= total) return;
    int node = i / H;
    int f    = i - node * H;
    float d  = dis[node];
    outp[i]  = h[i] * d * d + bias[f];
}

// ---------- edge scatter: agg[dst] += h[src] * dis[src]*dis[dst] ----------
// 64 threads per edge, float4 gather + 4 scalar atomics; node arrays (51 MB)
// are L2-resident (192 MB L2), so gather + atomics run at L2 bandwidth.
__global__ void k_edge_agg(const int* __restrict__ src, const int* __restrict__ dst,
                           const float* __restrict__ dis, const float* __restrict__ h,
                           float* __restrict__ agg, int E, int H)
{
    int idx    = blockIdx.x * blockDim.x + threadIdx.x;
    int chunks = H >> 2;                       // 64
    int total  = E * chunks;                   // 51.2M < 2^31
    if (idx >= total) return;
    int e  = idx >> 6;                         // chunks == 64
    int f4 = (idx & 63) << 2;
    int s  = src[e], d = dst[e];
    float coef = dis[s] * dis[d];
    const float4 hv = *(const float4*)(h + (size_t)s * H + f4);
    float* base = agg + (size_t)d * H + f4;
    atomicAdd(base + 0, hv.x * coef);
    atomicAdd(base + 1, hv.y * coef);
    atomicAdd(base + 2, hv.z * coef);
    atomicAdd(base + 3, hv.w * coef);
}

// ---------- pooling ----------
__global__ void k_cnt(const int* __restrict__ batch, float* __restrict__ cnt, int n) {
    int i = blockIdx.x * blockDim.x + threadIdx.x;
    if (i < n) atomicAdd(&cnt[batch[i]], 1.0f);
}

__global__ void k_pool(const float* __restrict__ h, const int* __restrict__ batch,
                       float* __restrict__ g, int total, int H)
{
    int i = blockIdx.x * blockDim.x + threadIdx.x;
    if (i >= total) return;
    int node = i / H;
    int f    = i - node * H;
    int b    = batch[node];
    float v  = h[i];
    atomicAdd(&g[(size_t)b * (2 * HDIM) + f], v);
    // post-ReLU v >= 0, so int-bit max == float max; init 0 matches empty-graph guard
    atomicMax((int*)&g[(size_t)b * (2 * HDIM) + HDIM + f], __float_as_int(v));
}

__global__ void k_pool_final(float* __restrict__ g, const float* __restrict__ cnt, int Gn) {
    int i = blockIdx.x * blockDim.x + threadIdx.x;
    if (i >= Gn * HDIM) return;
    int gi = i / HDIM;
    int f  = i - gi * HDIM;
    g[(size_t)gi * (2 * HDIM) + f] /= fmaxf(cnt[gi], 1.0f);
}

// ---------- head: out[64,5] = m2[64,256] @ Wc3[256,5] + bc3 ----------
__global__ void k_head(const float* __restrict__ m2, const float* __restrict__ Wc3,
                       const float* __restrict__ bc3, float* __restrict__ out)
{
    int i = blockIdx.x * blockDim.x + threadIdx.x;
    if (i >= N_GRAPH * 5) return;
    int gi = i / 5;
    int o  = i - gi * 5;
    float acc = bc3[o];
    const float* row = m2 + (size_t)gi * HDIM;
#pragma unroll 4
    for (int k = 0; k < HDIM; ++k) acc += row[k] * Wc3[k * 5 + o];
    out[i] = acc;
}

extern "C" void kernel_launch(void* const* d_in, const int* in_sizes, int n_in,
                              void* d_out, int out_size, void* d_ws, size_t ws_size,
                              hipStream_t stream) {
    (void)in_sizes; (void)n_in; (void)out_size; (void)ws_size;
    const float* x     = (const float*)d_in[0];
    const int*   ei    = (const int*)  d_in[1];
    const int*   batch = (const int*)  d_in[2];
    const float* W1    = (const float*)d_in[3];
    const float* b1    = (const float*)d_in[4];
    const float* W2    = (const float*)d_in[5];
    const float* b2    = (const float*)d_in[6];
    const float* W3    = (const float*)d_in[7];
    const float* b3    = (const float*)d_in[8];
    const float* Wc1   = (const float*)d_in[9];
    const float* bc1   = (const float*)d_in[10];
    const float* Wc2   = (const float*)d_in[11];
    const float* bc2   = (const float*)d_in[12];
    const float* Wc3   = (const float*)d_in[13];
    const float* bc3   = (const float*)d_in[14];
    const int* src  = ei;
    const int* dstp = ei + N_EDGES;

    // workspace layout (~105 MB)
    float* dis  = (float*)d_ws;                      // N
    float* bufA = dis  + N_NODES;                    // N*H (gemm output h)
    float* bufB = bufA + (size_t)N_NODES * HDIM;     // N*H (agg / layer input)
    float* g    = bufB + (size_t)N_NODES * HDIM;     // G*512
    float* cnt  = g    + N_GRAPH * 2 * HDIM;         // G
    float* m1   = cnt  + N_GRAPH;                    // G*512
    float* m2   = m1   + N_GRAPH * 2 * HDIM;         // G*256
    float* w1t  = m2   + N_GRAPH * HDIM;             // 256*128
    float* w2t  = w1t  + HDIM * F_IN;                // 256*256
    float* w3t  = w2t  + HDIM * HDIM;                // 256*256
    float* wc1t = w3t  + HDIM * HDIM;                // 512*512
    float* wc2t = wc1t + 512 * 512;                  // 256*512

    const int TPB = 256;
    const int NH  = N_NODES * HDIM;
    const dim3 gblk(32, 4);

    // pre-transpose all GEMM weights (constant work, ~2 MB total)
    k_transpose<<<(F_IN * HDIM + TPB - 1) / TPB, TPB, 0, stream>>>(W1,  w1t,  F_IN, HDIM);
    k_transpose<<<(HDIM * HDIM + TPB - 1) / TPB, TPB, 0, stream>>>(W2,  w2t,  HDIM, HDIM);
    k_transpose<<<(HDIM * HDIM + TPB - 1) / TPB, TPB, 0, stream>>>(W3,  w3t,  HDIM, HDIM);
    k_transpose<<<(512 * 512   + TPB - 1) / TPB, TPB, 0, stream>>>(Wc1, wc1t, 512, 512);
    k_transpose<<<(512 * 256   + TPB - 1) / TPB, TPB, 0, stream>>>(Wc2, wc2t, 512, 256);

    // degrees with self loop: deg=1 + sum over dst; dis = rsqrt(deg)
    k_fill<<<(N_NODES + TPB - 1) / TPB, TPB, 0, stream>>>(dis, 1.0f, N_NODES);
    k_deg<<<(N_EDGES + TPB - 1) / TPB, TPB, 0, stream>>>(dstp, dis, N_EDGES);
    k_rsqrt<<<(N_NODES + TPB - 1) / TPB, TPB, 0, stream>>>(dis, N_NODES);

    auto gcn_layer = [&](const float* in, int K, const float* Wt, const float* bb) {
        int strips = (N_NODES / 16) * (HDIM / 64);          // 3125 * 4 = 12500
        k_gemm_wmma<false, false><<<(strips + 3) / 4, gblk, 0, stream>>>(
            in, Wt, bufA, N_NODES, K, HDIM, nullptr);
        k_init_agg<<<(NH + TPB - 1) / TPB, TPB, 0, stream>>>(bufA, dis, bb, bufB, NH, HDIM);
        int etot = N_EDGES * (HDIM / 4);
        k_edge_agg<<<(etot + TPB - 1) / TPB, TPB, 0, stream>>>(src, dstp, dis, bufA,
                                                               bufB, N_EDGES, HDIM);
        k_relu<<<(NH + TPB - 1) / TPB, TPB, 0, stream>>>(bufB, NH);
    };

    gcn_layer(x,    F_IN, w1t, b1);   // layer 1: reads x,    result in bufB
    gcn_layer(bufB, HDIM, w2t, b2);   // layer 2: reads bufB, gemm->bufA, result in bufB
    gcn_layer(bufB, HDIM, w3t, b3);   // layer 3

    // global mean/max pool per graph -> g[G, 512]
    k_fill<<<(N_GRAPH * 2 * HDIM + TPB - 1) / TPB, TPB, 0, stream>>>(g, 0.0f, N_GRAPH * 2 * HDIM);
    k_fill<<<1, N_GRAPH, 0, stream>>>(cnt, 0.0f, N_GRAPH);
    k_cnt<<<(N_NODES + TPB - 1) / TPB, TPB, 0, stream>>>(batch, cnt, N_NODES);
    k_pool<<<(NH + TPB - 1) / TPB, TPB, 0, stream>>>(bufB, batch, g, NH, HDIM);
    k_pool_final<<<(N_GRAPH * HDIM + TPB - 1) / TPB, TPB, 0, stream>>>(g, cnt, N_GRAPH);

    // MLP head via WMMA (bias + relu fused in epilogue)
    {
        int s1 = (N_GRAPH / 16) * (512 / 64);   // 32 strips
        k_gemm_wmma<true, true><<<(s1 + 3) / 4, gblk, 0, stream>>>(
            g, wc1t, m1, N_GRAPH, 512, 512, bc1);
        int s2 = (N_GRAPH / 16) * (256 / 64);   // 16 strips
        k_gemm_wmma<true, true><<<(s2 + 3) / 4, gblk, 0, stream>>>(
            m1, wc2t, m2, N_GRAPH, 512, 256, bc2);
    }
    k_head<<<(N_GRAPH * 5 + TPB - 1) / TPB, TPB, 0, stream>>>(m2, Wc3, bc3, (float*)d_out);
}
